// InterModule_12756052869768
// MI455X (gfx1250) — matrix operand order repaired and checked by reading, not compiled, not run on hardware
//
#include <hip/hip_runtime.h>
#include <math.h>
#include <stdint.h>

// ---------------------------------------------------------------------------
// CDNA5 / gfx1250, wave32. Matrix engine: v_wmma_f32_16x16x32_bf16.
// Fragment layouts assumed per CDNA5 ISA §7.12.2:
//   A (16x32 bf16): lane L -> row m = L&15; khalf = L>>4;
//       elements e=0..15 -> k = (e&8 ? 16:0) + khalf*8 + (e&7)
//       => two contiguous 8-element (16B) chunks at k = khalf*8 and k = 16+khalf*8
//   B (32x16 bf16): lane L -> col n = L&15; kh = L>>4; element e -> k = (e&8 ? 16:0) + kh*8 + (e&7)
//       => we pre-pack weights so each lane reads its 16 elems as 32 contiguous bytes
//   C/D (16x16 f32, 8 VGPRs): VGPR r, lane L -> row = r + 8*(L>>4), col = L&15
// ---------------------------------------------------------------------------

typedef __attribute__((ext_vector_type(16))) __bf16 v16bf;
typedef __attribute__((ext_vector_type(8)))  float  v8f;

union Frag {
  v16bf v;
  uint4 q[2];
};

__device__ inline v8f vzero8() {
  v8f z;
#pragma unroll
  for (int i = 0; i < 8; ++i) z[i] = 0.0f;
  return z;
}

// ---------------------------------------------------------------------------
// Pack a row-major [K,N] f32 weight into bf16 WMMA-B fragment order:
// frag f = nt*Kt + kt ; within frag: [lane][e] contiguous (32B per lane).
// ---------------------------------------------------------------------------
__global__ void pack_weights_kernel(const float* __restrict__ w,
                                    __bf16* __restrict__ dst, int K, int N) {
  int i = blockIdx.x * blockDim.x + threadIdx.x;
  if (i >= K * N) return;
  int k = i / N;
  int n = i - k * N;
  int Kt = K >> 5;
  int kt = k >> 5;
  int kr = k & 31;
  int kh = (kr >> 3) & 1;
  int e  = (kr & 7) | ((kr >> 4) << 3);
  int nt = n >> 4;
  int nc = n & 15;
  int lane = (kh << 4) | nc;
  size_t o = (((size_t)(nt * Kt + kt) * 32) + lane) * 16 + e;
  dst[o] = (__bf16)w[i];
}

// ---------------------------------------------------------------------------
// Fused per-l chain: out_l = (1/sqrt(MID)) * W2^T @ [ normact( (1/sqrt(IN)) * W1^T @ x_l ) ]
// One workgroup = 16 samples. 8 waves split output column tiles.
// NormActivation done in registers across the D component accumulators:
//   silu(n)/n == sigmoid(n).
// ---------------------------------------------------------------------------
template <int MUL_IN, int MUL_MID, int MUL_OUT, int D, int OFF_IN, int OFF_OUT>
__global__ __launch_bounds__(256) void chain_kernel(
    const float* __restrict__ x, const __bf16* __restrict__ w1p,
    const __bf16* __restrict__ w2p, float* __restrict__ out) {
  constexpr int SA  = MUL_IN + 8;   // +8 bf16 pad: break 64-bank periodicity, keep 16B align
  constexpr int SH  = MUL_MID + 8;
  constexpr int KT1 = MUL_IN / 32,  NT1 = MUL_MID / 16;
  constexpr int KT2 = MUL_MID / 32, NT2 = MUL_OUT / 16;

  extern __shared__ __bf16 smem[];
  __bf16* xp = smem;                     // D planes of [16][SA]
  __bf16* hp = smem + D * 16 * SA;       // D planes of [16][SH]

  const int tid  = threadIdx.x;
  const int wave = tid >> 5;
  const int lane = tid & 31;
  const int m    = lane & 15;
  const int kh   = lane >> 4;
  const int n0   = blockIdx.x * 16;

  // ---- stage x tile: f32 global -> bf16 LDS, de-interleave components ----
  const float* xrow = x + (size_t)n0 * 1920 + OFF_IN;
  for (int i = tid; i < 16 * MUL_IN * D; i += 256) {   // multiple of 256: no divergence
    int row = i / (MUL_IN * D);
    int col = i - row * (MUL_IN * D);
    int u = col / D, c = col - u * D;
    xp[(c * 16 + row) * SA + u] = (__bf16)xrow[(size_t)row * 1920 + col];
  }
  __syncthreads();

  // ---- linear1 + norm-act (register resident) ----
  const float s1 = rsqrtf((float)MUL_IN);
  for (int vt = wave; vt < NT1; vt += 8) {
    v8f acc[D];
#pragma unroll
    for (int c = 0; c < D; ++c) acc[c] = vzero8();
    for (int kt = 0; kt < KT1; ++kt) {
      Frag b;  // shared across components
      const uint4* bp =
          (const uint4*)(w1p + (((size_t)(vt * KT1 + kt) * 32) + lane) * 16);
      b.q[0] = bp[0];
      b.q[1] = bp[1];
#pragma unroll
      for (int c = 0; c < D; ++c) {
        Frag a;
        const __bf16* ap = xp + (c * 16 + m) * SA + kt * 32 + kh * 8;
        a.q[0] = *(const uint4*)(ap);
        a.q[1] = *(const uint4*)(ap + 16);
        acc[c] = __builtin_amdgcn_wmma_f32_16x16x32_bf16(
            false, a.v, false, b.v, (short)0, acc[c], false, false);
      }
    }
#pragma unroll
    for (int r = 0; r < 8; ++r) {
      float ss = 0.0f;
#pragma unroll
      for (int c = 0; c < D; ++c) {
        float y = acc[c][r] * s1;
        ss += y * y;
      }
      float nn = sqrtf(ss);
      float sc = s1 / (1.0f + __expf(-nn));  // s1 * sigmoid(||y||) == s1*silu(n)/n
      int sample = r + (kh << 3);
#pragma unroll
      for (int c = 0; c < D; ++c)
        hp[(c * 16 + sample) * SH + vt * 16 + m] = (__bf16)(acc[c][r] * sc);
    }
  }
  __syncthreads();

  // ---- linear2, store f32 directly to global ----
  const float s2 = rsqrtf((float)MUL_MID);
  float* orow = out + (size_t)n0 * 1920 + OFF_OUT;
  for (int vt = wave; vt < NT2; vt += 8) {
    v8f acc[D];
#pragma unroll
    for (int c = 0; c < D; ++c) acc[c] = vzero8();
    for (int kt = 0; kt < KT2; ++kt) {
      Frag b;
      const uint4* bp =
          (const uint4*)(w2p + (((size_t)(vt * KT2 + kt) * 32) + lane) * 16);
      b.q[0] = bp[0];
      b.q[1] = bp[1];
#pragma unroll
      for (int c = 0; c < D; ++c) {
        Frag a;
        const __bf16* ap = hp + (c * 16 + m) * SH + kt * 32 + kh * 8;
        a.q[0] = *(const uint4*)(ap);
        a.q[1] = *(const uint4*)(ap + 16);
        acc[c] = __builtin_amdgcn_wmma_f32_16x16x32_bf16(
            false, a.v, false, b.v, (short)0, acc[c], false, false);
      }
    }
#pragma unroll
    for (int r = 0; r < 8; ++r) {
      int sample = r + (kh << 3);
#pragma unroll
      for (int c = 0; c < D; ++c)
        orow[(size_t)sample * 1920 + (vt * 16 + m) * D + c] = acc[c][r] * s2;
    }
  }
}

// ---------------------------------------------------------------------------
// Packed-weight element offsets inside d_ws (bf16 elements)
// ---------------------------------------------------------------------------
static const size_t W1L0 = 0;                    // 512*1024
static const size_t W1L1 = W1L0 + 512 * 1024;    // 256*512
static const size_t W1L2 = W1L1 + 256 * 512;     // 128*256
static const size_t W2L0 = W1L2 + 128 * 256;     // 1024*512
static const size_t W2L1 = W2L0 + 1024 * 512;    // 512*256
static const size_t W2L2 = W2L1 + 512 * 256;     // 256*128

extern "C" void kernel_launch(void* const* d_in, const int* in_sizes, int n_in,
                              void* d_out, int out_size, void* d_ws, size_t ws_size,
                              hipStream_t stream) {
  (void)in_sizes; (void)n_in; (void)out_size; (void)ws_size;
  const float* xf = (const float*)d_in[0];
  float* outf = (float*)d_out;
  __bf16* wp = (__bf16*)d_ws;

  // 1) repack weights f32 -> bf16 WMMA-B fragment order (tiny, L2-resident)
  struct { int idx; size_t off; int K, N; } packs[6] = {
      {1, W1L0, 512, 1024}, {2, W1L1, 256, 512}, {3, W1L2, 128, 256},
      {4, W2L0, 1024, 512}, {5, W2L1, 512, 256}, {6, W2L2, 256, 128}};
  for (int p = 0; p < 6; ++p) {
    int total = packs[p].K * packs[p].N;
    pack_weights_kernel<<<(total + 255) / 256, 256, 0, stream>>>(
        (const float*)d_in[packs[p].idx], wp + packs[p].off, packs[p].K, packs[p].N);
  }

  // 2) three independent fused l-chains, 2048 workgroups x 256 threads each
  dim3 grid(32768 / 16), block(256);
  const int smem0 = (1 * 16 * (512 + 8) + 1 * 16 * (1024 + 8)) * 2;  // 49664 B
  const int smem1 = (3 * 16 * (256 + 8) + 3 * 16 * (512 + 8)) * 2;   // 75264 B
  const int smem2 = (5 * 16 * (128 + 8) + 5 * 16 * (256 + 8)) * 2;   // 64000 B

  chain_kernel<512, 1024, 512, 1, 0, 0>
      <<<grid, block, smem0, stream>>>(xf, wp + W1L0, wp + W2L0, outf);
  chain_kernel<256, 512, 256, 3, 512, 512>
      <<<grid, block, smem1, stream>>>(xf, wp + W1L1, wp + W2L1, outf);
  chain_kernel<128, 256, 128, 5, 1280, 1280>
      <<<grid, block, smem2, stream>>>(xf, wp + W1L2, wp + W2L2, outf);
}
